// SNNModel3_86852828659875
// MI455X (gfx1250) — matrix operand order repaired and checked by reading, not compiled, not run on hardware
//
#include <hip/hip_runtime.h>

typedef __attribute__((ext_vector_type(16))) _Float16 v16h;
typedef __attribute__((ext_vector_type(8)))  float    v8f;

#define S_TOT   131072
#define D_DIM   96
#define NSTEPS  7
#define STEP_   7
#define BIN_    50
#define BETA_   0.5f
#define THR_    0.5f
#define EPS_    1e-5f

#define WAVES_PER_BLOCK 2
#define ROWS_PER_WAVE   16
#define ROWS_PER_BLOCK  (WAVES_PER_BLOCK * ROWS_PER_WAVE)   // 32
#define HIST  49                                            // history rows (BIN-2+1)
#define CSLEN (ROWS_PER_WAVE + HIST + 1)                    // 66 cumsum entries

__global__ __launch_bounds__(WAVES_PER_BLOCK * 32)
void snn_kernel(const float* __restrict__ x,   const float* __restrict__ ln_w,
                const float* __restrict__ ln_b, const float* __restrict__ w1,
                const float* __restrict__ b1,   const float* __restrict__ w2,
                const float* __restrict__ b2,   const float* __restrict__ w3,
                const float* __restrict__ b3,   const float* __restrict__ v_x,
                const float* __restrict__ v_y,  float* __restrict__ out)
{
    // ---- LDS: 2*25344 + 2*1024 + 2*1536 + 2688 + 2688 + 384 = 61568 B ----
    __shared__ float    cs  [WAVES_PER_BLOCK][CSLEN * D_DIM];   // windowed prefix sums (f32)
    __shared__ _Float16 s1l [WAVES_PER_BLOCK][16 * 32];         // spike layer-1 (C->A relayout)
    __shared__ _Float16 s2l [WAVES_PER_BLOCK][16 * 48];         // spike layer-2
    __shared__ float    lnw_s[NSTEPS * D_DIM];
    __shared__ float    lnb_s[NSTEPS * D_DIM];
    __shared__ float    w3_s[2 * 48];

    const int tid  = threadIdx.x;
    const int wave = tid >> 5;
    const int lane = tid & 31;
    const int lid  = lane & 15;       // M index / N index within 16-tile
    const int hi   = lane >> 4;       // half-wave selector

    // ---- preload LN params + w3 into LDS (shared across waves) ----
    for (int i = tid; i < NSTEPS * D_DIM; i += WAVES_PER_BLOCK * 32) {
        lnw_s[i] = ln_w[i];
        lnb_s[i] = ln_b[i];
    }
    for (int i = tid; i < 2 * 48; i += WAVES_PER_BLOCK * 32) w3_s[i] = w3[i];
    __syncthreads();

    const int t0   = blockIdx.x * ROWS_PER_BLOCK + wave * ROWS_PER_WAVE;
    const int base = t0 - HIST;                 // first xs row of cs window (may be < 0)
    float* csW = cs[wave];

    // ---- per-wave local cumulative sums over the history window (f32) ----
    // cs[j][d] = sum of x rows [base, base+j-1] (rows < 0 contribute 0)
    #pragma unroll
    for (int u = 0; u < 3; ++u) {
        const int d = lane + 32 * u;
        float run = 0.f;
        csW[0 * D_DIM + d] = 0.f;
        for (int j = 1; j < CSLEN; ++j) {
            const int g = base + j - 1;
            const float v = (g >= 0) ? x[g * D_DIM + d] : 0.f;
            run += v;
            csW[j * D_DIM + d] = run;
        }
    }
    // same-wave LDS ops are in-order: no barrier needed before reading csW

    // ---- per-row window bounds (jl/jh index into csW) ----
    const int t  = t0 + lid;
    const int st = (t - (BIN_ - 2) > 0) ? (t - (BIN_ - 2)) : 0;
    const int en = t + 2;
    int jl[NSTEPS], jh[NSTEPS];
    #pragma unroll
    for (int i = 0; i < NSTEPS; ++i) {
        int lo = st + STEP_ * i; if (lo > en) lo = en;
        int h  = lo + STEP_;     if (h  > en) h  = en;
        int m0 = lo - 1;         if (m0 < 0)  m0 = 0;
        int a  = m0 - base;
        int b  = (h - 2) + 1 - base;
        if (b < a) b = a;        // empty window -> zero sum
        jl[i] = a; jh[i] = b;
    }

    // ---- LayerNorm stats over the 7x96 block of this row ----
    float sum = 0.f, ssq = 0.f;
    const int d0 = hi * 48;      // lane pair splits the 96 dims
    #pragma unroll
    for (int i = 0; i < NSTEPS; ++i) {
        const float* ch = &csW[jh[i] * D_DIM + d0];
        const float* cl = &csW[jl[i] * D_DIM + d0];
        for (int dd = 0; dd < 48; ++dd) {
            const float v = ch[dd] - cl[dd];
            sum += v; ssq += v * v;
        }
    }
    sum += __shfl_xor(sum, 16, 32);
    ssq += __shfl_xor(ssq, 16, 32);
    const float invn = 1.f / (float)(NSTEPS * D_DIM);
    const float mu   = sum * invn;
    float       var  = ssq * invn - mu * mu;
    if (var < 0.f) var = 0.f;
    const float rstd = rsqrtf(var + EPS_);

    // ---- preload weights into WMMA B-layout registers (f16) ----
    // B layout (32x16, 8 VGPRs): lane -> N = lid, element e -> K = 8*hi + (e<8 ? e : e+8)
    const int abase8 = hi * 8;   // A layout: element e -> K = abase8 + (e<8 ? e : e+8)
    v16h B1[2][3], B2[3];
    #pragma unroll
    for (int nt = 0; nt < 2; ++nt) {
        const int n = nt * 16 + lid;
        #pragma unroll
        for (int c = 0; c < 3; ++c)
            #pragma unroll
            for (int e = 0; e < 16; ++e)
                B1[nt][c][e] = (_Float16)w1[n * D_DIM + 32 * c + 8 * hi + (e < 8 ? e : e + 8)];
    }
    #pragma unroll
    for (int nt = 0; nt < 3; ++nt) {
        const int n = nt * 16 + lid;
        #pragma unroll
        for (int e = 0; e < 16; ++e)
            B2[nt][e] = (_Float16)w2[n * 32 + 8 * hi + (e < 8 ? e : e + 8)];
    }
    const float b1n[2] = { b1[lid], b1[16 + lid] };
    const float b2n[3] = { b2[lid], b2[16 + lid], b2[32 + lid] };
    const float b3c = b3[hi];
    const float vsc = hi ? v_y[0] : v_x[0];

    // ---- SNN scan: membrane state lives in the C/D register layout ----
    v8f m1t[2] = {};            // (16x32) layer-1 membrane
    v8f m2t[3] = {};            // (16x48) layer-2 membrane
    float m3 = 0.f;             // this lane's (row=lid, col=hi) layer-3 membrane
    _Float16* s1w = s1l[wave];
    _Float16* s2w = s2l[wave];

    #pragma unroll
    for (int i = 0; i < NSTEPS; ++i) {
        // A operand for layer 1: normalized window sums, 3 K-chunks of 32
        v16h A1[3];
        #pragma unroll
        for (int c = 0; c < 3; ++c)
            #pragma unroll
            for (int e = 0; e < 16; ++e) {
                const int k = abase8 + (e < 8 ? e : e + 8);
                const int d = 32 * c + k;
                const float raw = csW[jh[i] * D_DIM + d] - csW[jl[i] * D_DIM + d];
                const float xv  = (raw - mu) * rstd * lnw_s[i * D_DIM + d]
                                  + lnb_s[i * D_DIM + d];
                A1[c][e] = (_Float16)xv;
            }

        // layer 1: cur1 = xn @ w1.T + b1  (two 16x16 N-tiles, K=96)
        #pragma unroll
        for (int nt = 0; nt < 2; ++nt) {
            v8f acc;
            #pragma unroll
            for (int j = 0; j < 8; ++j) acc[j] = b1n[nt];
            #pragma unroll
            for (int c = 0; c < 3; ++c)
                acc = __builtin_amdgcn_wmma_f32_16x16x32_f16(
                        false, A1[c], false, B1[nt][c], (short)0, acc, false, false);
            // membrane update + spike, store s1 in C layout to LDS
            #pragma unroll
            for (int j = 0; j < 8; ++j) {
                const float old = m1t[nt][j];
                const float r1  = (old > THR_) ? THR_ : 0.f;   // r1 * THR
                const float nv  = BETA_ * old + acc[j] - r1;
                m1t[nt][j] = nv;
                const float s = (nv > THR_) ? 1.f : 0.f;
                const int M = j + 8 * hi;
                s1w[M * 32 + nt * 16 + lid] = (_Float16)s;
            }
        }

        // relayout s1 (C layout in LDS) -> A layout registers
        v16h A2;
        #pragma unroll
        for (int e = 0; e < 16; ++e)
            A2[e] = s1w[lid * 32 + abase8 + (e < 8 ? e : e + 8)];

        // layer 2: cur2 = s1 @ w2.T + b2  (three N-tiles, K=32)
        #pragma unroll
        for (int nt = 0; nt < 3; ++nt) {
            v8f acc;
            #pragma unroll
            for (int j = 0; j < 8; ++j) acc[j] = b2n[nt];
            acc = __builtin_amdgcn_wmma_f32_16x16x32_f16(
                    false, A2, false, B2[nt], (short)0, acc, false, false);
            #pragma unroll
            for (int j = 0; j < 8; ++j) {
                const float old = m2t[nt][j];
                const float r2  = (old > THR_) ? THR_ : 0.f;
                const float nv  = BETA_ * old + acc[j] - r2;
                m2t[nt][j] = nv;
                const float s = (nv > THR_) ? 1.f : 0.f;
                const int M = j + 8 * hi;
                s2w[M * 48 + nt * 16 + lid] = (_Float16)s;
            }
        }

        // layer 3 (N=2): one 48-wide dot per lane; lane = (row=lid, col=hi)
        float dot = b3c;
        #pragma unroll
        for (int k = 0; k < 48; ++k)
            dot += (float)s2w[lid * 48 + k] * w3_s[hi * 48 + k];
        m3 = BETA_ * m3 + dot;
    }

    out[t * 2 + hi] = vsc * m3;
}

extern "C" void kernel_launch(void* const* d_in, const int* in_sizes, int n_in,
                              void* d_out, int out_size, void* d_ws, size_t ws_size,
                              hipStream_t stream)
{
    const float* x   = (const float*)d_in[0];
    const float* lnw = (const float*)d_in[1];
    const float* lnb = (const float*)d_in[2];
    const float* w1  = (const float*)d_in[3];
    const float* b1  = (const float*)d_in[4];
    const float* w2  = (const float*)d_in[5];
    const float* b2  = (const float*)d_in[6];
    const float* w3  = (const float*)d_in[7];
    const float* b3  = (const float*)d_in[8];
    const float* vx  = (const float*)d_in[9];
    const float* vy  = (const float*)d_in[10];
    float* out = (float*)d_out;

    dim3 grid(S_TOT / ROWS_PER_BLOCK);     // 4096 blocks
    dim3 block(WAVES_PER_BLOCK * 32);      // 64 threads = 2 wave32
    hipLaunchKernelGGL(snn_kernel, grid, block, 0, stream,
                       x, lnw, lnb, w1, b1, w2, b2, w3, b3, vx, vy, out);
}
